// SelfAttention_54245436949089
// MI455X (gfx1250) — matrix operand order repaired and checked by reading, not compiled, not run on hardware
//
#include <hip/hip_runtime.h>
#include <hip/hip_bf16.h>

typedef __attribute__((ext_vector_type(16))) _Float16 v16h;
typedef __attribute__((ext_vector_type(8)))  _Float16 v8h;
typedef __attribute__((ext_vector_type(8)))  float    v8f;
typedef __attribute__((ext_vector_type(4)))  float    v4f;
typedef __attribute__((ext_vector_type(4)))  int      v4i;

#define BB 2
#define NSEQ 2048
#define CDIM 1024
#define NHEADS 16
#define HD 64
#define RLORA 4
#define LORA_SCALING 0.25f
#define MROWS (BB * NSEQ)      /* 4096 */
#define OUTQKV (3 * CDIM)      /* 3072 */

// ---------------------------------------------------------------------------
// gfx1250 async global->LDS path (probe-confirmed present in this toolchain):
//   void __builtin_amdgcn_global_load_async_to_lds_b128(
//       int4 addrspace(1)* src, int4 addrspace(3)* dst, imm offset, imm cpol)
// ---------------------------------------------------------------------------
#if defined(__has_builtin)
#if __has_builtin(__builtin_amdgcn_global_load_async_to_lds_b128) && \
    __has_builtin(__builtin_amdgcn_s_wait_asynccnt)
#define USE_ASYNC_LDS 1
#endif
#endif
#ifndef USE_ASYNC_LDS
#define USE_ASYNC_LDS 0
#endif

#if USE_ASYNC_LDS
typedef __attribute__((address_space(1))) v4i glb_v4i;
typedef __attribute__((address_space(3))) v4i lds_v4i;
#endif

// Copy 16 bytes global -> LDS (async if available).
__device__ __forceinline__ void cp16_g2l(void* lds_dst, const void* gsrc) {
#if USE_ASYNC_LDS
  __builtin_amdgcn_global_load_async_to_lds_b128(
      (glb_v4i*)(uintptr_t)gsrc,
      (lds_v4i*)(unsigned)(uintptr_t)lds_dst, 0, 0);
#else
  *(v8h*)lds_dst = *(const v8h*)gsrc;
#endif
}
__device__ __forceinline__ void async_wait_all() {
#if USE_ASYNC_LDS
  __builtin_amdgcn_s_wait_asynccnt(0);
#endif
}

// ---------------------------------------------------------------------------
// WMMA fragment helpers (contiguous k = 16*g + e in both A and B, wave32)
// ---------------------------------------------------------------------------

// A fragment, 16x32 f16, source row-major (global or LDS).
// lane l: M = l&15, group g = l>>4; element e -> K = g*16 + e
__device__ __forceinline__ v16h load_A_frag(const _Float16* __restrict__ base,
                                            int lda, int row0, int k0, int lane) {
  const int g = lane >> 4, m = lane & 15;
  const _Float16* p = base + (size_t)(row0 + m) * lda + k0 + g * 16;
  v8h lo = *(const v8h*)(p);
  v8h hi = *(const v8h*)(p + 8);
  v16h r;
#pragma unroll
  for (int i = 0; i < 8; ++i) { r[i] = lo[i]; r[i + 8] = hi[i]; }
  return r;
}

// B fragment, 32x16 f16, logical B[k][n] = Src[n0+n][k0+k] (Src row-major).
// lane l: N = l&15, group g = l>>4; element e -> K = g*16 + e => one 32B load.
__device__ __forceinline__ v16h load_B_frag(const _Float16* __restrict__ base,
                                            int ld, int n0, int k0, int lane) {
  const int g = lane >> 4, n = lane & 15;
  return *(const v16h*)(base + (size_t)(n0 + n) * ld + k0 + g * 16);
}

__device__ __forceinline__ v8f wmma_f16(v16h a, v16h b, v8f c) {
  return __builtin_amdgcn_wmma_f32_16x16x32_f16(false, a, false, b,
                                                (short)0, c, false, false);
}

// ---------------------------------------------------------------------------
// Kernel 0: f32 -> f16 conversion (grid-stride)
// ---------------------------------------------------------------------------
__global__ void cvt_f32_to_f16(const float* __restrict__ in,
                               _Float16* __restrict__ out, int n) {
  for (int i = blockIdx.x * blockDim.x + threadIdx.x; i < n;
       i += gridDim.x * blockDim.x)
    out[i] = (_Float16)in[i];
}

// ---------------------------------------------------------------------------
// Kernel 1: LoRA a = x . lora_A^T  -> abuf[MROWS][8] f32 (tiny, scalar)
// ---------------------------------------------------------------------------
__global__ void lora_a_kernel(const float* __restrict__ x,
                              const float* __restrict__ lora_A,
                              float* __restrict__ abuf) {
  int t = blockIdx.x * blockDim.x + threadIdx.x;
  if (t >= MROWS * 2 * RLORA) return;
  int r = t & 7, m = t >> 3;
  const float* xr = x + (size_t)m * CDIM;
  const float* ar = lora_A + (size_t)r * CDIM;
  float s = 0.f;
  for (int c = 0; c < CDIM; ++c) s += xr[c] * ar[c];
  abuf[t] = s;
}

// ---------------------------------------------------------------------------
// Kernel 2: fused QKV GEMM + bias + LoRA epilogue.
// Block = 128 threads (4 waves); each wave a 64x64 macro-tile (4x4 WMMA
// tiles). The 64x32 weight tile is staged in LDS with DOUBLE-BUFFERED async
// copies (copy of k0+32 overlaps the 16 WMMAs of k0). The destination
// (q/k/v + head) is wave-uniform since o0 is a multiple of 64.
// ---------------------------------------------------------------------------
__global__ void __launch_bounds__(128) qkv_wmma_kernel(
    const _Float16* __restrict__ xh, const _Float16* __restrict__ wqkv_h,
    const float* __restrict__ b_qkv, const float* __restrict__ lora_B,
    const float* __restrict__ abuf, _Float16* __restrict__ qbuf,
    _Float16* __restrict__ kbuf, _Float16* __restrict__ vtbuf) {
  __shared__ __align__(32) _Float16 btile[2][64 * 32];  // 2 x 4KB
  const int tid = threadIdx.x, lane = tid & 31, wid = tid >> 5;
  const int o0 = blockIdx.x * 64;                 // output-channel block
  const int mbase = blockIdx.y * 256 + wid * 64;  // row block
  const int g = lane >> 4, n = lane & 15;

  auto stage = [&](int k0, int buf) {
#pragma unroll
    for (int r = 0; r < 2; ++r) {
      int c = tid + 128 * r;
      int row = c >> 2, sub = c & 3;
      cp16_g2l(&btile[buf][row * 32 + sub * 8],
               wqkv_h + (size_t)(o0 + row) * CDIM + k0 + sub * 8);
    }
  };

  v8f acc[4][4] = {};
  stage(0, 0);
  async_wait_all();
  __syncthreads();
  for (int k0 = 0; k0 < CDIM; k0 += 32) {
    const int cur = (k0 >> 5) & 1;
    if (k0 + 32 < CDIM) stage(k0 + 32, cur ^ 1);  // overlap copy with compute

    v16h af[4], bf[4];
#pragma unroll
    for (int i = 0; i < 4; ++i)
      af[i] = load_A_frag(xh, CDIM, mbase + 16 * i, k0, lane);
#pragma unroll
    for (int j = 0; j < 4; ++j)
      bf[j] = load_B_frag(btile[cur], 32, 16 * j, 0, lane);
#pragma unroll
    for (int i = 0; i < 4; ++i)
#pragma unroll
      for (int j = 0; j < 4; ++j)
        acc[i][j] = wmma_f16(af[i], bf[j], acc[i][j]);

    async_wait_all();  // next buffer staged
    __syncthreads();   // all waves done with btile[cur]
  }

  // ---- Epilogue: wave-uniform destination (64-wide block = one of q/k/v,
  // one head; 64-row block never crosses the batch boundary). ----
  const int sel = o0 >> 10;        // 0=q, 1=k, 2=v
  const int h = (o0 & 1023) >> 6;  // head
  const int bi = mbase >> 11;      // batch index (uniform per wave)
  const int nseq0 = mbase & (NSEQ - 1);

#pragma unroll
  for (int jt = 0; jt < 4; ++jt) {
    const int o = o0 + 16 * jt + n;
    const float bias = b_qkv[o];
    float lb0 = 0.f, lb1 = 0.f, lb2 = 0.f, lb3 = 0.f;
    const int aoff = (sel == 2) ? RLORA : 0;
    if (sel != 1) {
      int ro = (sel == 0) ? o : (o - CDIM);  // lora_B rows: q:[0,C) v:[C,2C)
      lb0 = lora_B[ro * 4 + 0]; lb1 = lora_B[ro * 4 + 1];
      lb2 = lora_B[ro * 4 + 2]; lb3 = lora_B[ro * 4 + 3];
    }
#pragma unroll
    for (int it = 0; it < 4; ++it) {
      const int mloc = 16 * it + 8 * g;  // local row start for this lane
      const float* ab = abuf + (size_t)(mbase + mloc) * 8 + aoff;
      if (sel == 2) {
        // v: 8 consecutive nseq per lane -> one packed 16B store into vT
        v8h pk;
#pragma unroll
        for (int jj = 0; jj < 8; ++jj) {
          v4f av = *(const v4f*)(ab + jj * 8);
          float v = acc[it][jt][jj] + bias +
                    LORA_SCALING * (av.x * lb0 + av.y * lb1 + av.z * lb2 + av.w * lb3);
          pk[jj] = (_Float16)v;
        }
        const int d = 16 * jt + n;
        *(v8h*)(vtbuf + (((size_t)bi * NHEADS + h) * HD + d) * NSEQ + nseq0 + mloc) = pk;
      } else if (sel == 0) {
        _Float16* dst = qbuf +
            (((size_t)bi * NHEADS + h) * NSEQ + nseq0 + mloc) * HD + 16 * jt + n;
#pragma unroll
        for (int jj = 0; jj < 8; ++jj) {
          v4f av = *(const v4f*)(ab + jj * 8);
          float v = acc[it][jt][jj] + bias +
                    LORA_SCALING * (av.x * lb0 + av.y * lb1 + av.z * lb2 + av.w * lb3);
          dst[(size_t)jj * HD] = (_Float16)(v * 0.125f);  // fold softmax scale
        }
      } else {
        _Float16* dst = kbuf +
            (((size_t)bi * NHEADS + h) * NSEQ + nseq0 + mloc) * HD + 16 * jt + n;
#pragma unroll
        for (int jj = 0; jj < 8; ++jj)
          dst[(size_t)jj * HD] = (_Float16)(acc[it][jt][jj] + bias);
      }
    }
  }
}

// ---------------------------------------------------------------------------
// Kernel 3: flash attention. Block = 4 waves on the SAME (b,h) with 4
// consecutive 16-query blocks; the 32-key K and V tiles are double-buffered
// in LDS via async copies and shared by all 4 waves.
// ---------------------------------------------------------------------------
__global__ void __launch_bounds__(128) flash_attn_kernel(
    const _Float16* __restrict__ qbuf, const _Float16* __restrict__ kbuf,
    const _Float16* __restrict__ vtbuf, _Float16* __restrict__ aout) {
  __shared__ __align__(32) _Float16 ktile[2][32 * 64];    // keys x d, 2x4KB
  __shared__ __align__(32) _Float16 vtile[2][64 * 32];    // d x keys, 2x4KB
  __shared__ __align__(32) _Float16 pshare[4 * 16 * 32];  // per-wave P, 4KB

  const int tid = threadIdx.x, lane = tid & 31, wid = tid >> 5;
  const int bh = blockIdx.x >> 5;                // [0,32)
  const int qblk = (blockIdx.x & 31) * 4 + wid;  // [0,128)
  const int m0 = qblk * 16;
  const _Float16* qb = qbuf + (size_t)bh * NSEQ * HD;
  const _Float16* kb = kbuf + (size_t)bh * NSEQ * HD;
  const _Float16* vt = vtbuf + (size_t)bh * HD * NSEQ;
  _Float16* pb = pshare + wid * (16 * 32);
  const int g = lane >> 4, n = lane & 15;

  auto stage_kv = [&](int key0, int buf) {
#pragma unroll
    for (int r = 0; r < 2; ++r) {
      int c = tid + 128 * r;
      // K tile: contiguous 4KB slab of kbuf
      cp16_g2l(((char*)ktile[buf]) + c * 16,
               ((const char*)(kb + (size_t)key0 * HD)) + c * 16);
      // V tile: 64 rows x 64B gathered from transposed V
      int row = c >> 2, sub = c & 3;
      cp16_g2l(&vtile[buf][row * 32 + sub * 8],
               vt + (size_t)row * NSEQ + key0 + sub * 8);
    }
  };

  const v16h qf0 = load_A_frag(qb, HD, m0, 0, lane);
  const v16h qf1 = load_A_frag(qb, HD, m0, 32, lane);

  float mrun[8], lrun[8];
  v8f accO[4] = {};
#pragma unroll
  for (int j = 0; j < 8; ++j) { mrun[j] = -1e30f; lrun[j] = 0.f; }

  stage_kv(0, 0);
  async_wait_all();
  __syncthreads();
  for (int key0 = 0; key0 < NSEQ; key0 += 32) {
    const int cur = (key0 >> 5) & 1;
    if (key0 + 32 < NSEQ) stage_kv(key0 + 32, cur ^ 1);  // overlap copy

    // S = (Q*0.125) K^T : two 16x16 key tiles, hd=64 => 2 WMMAs each
    v8f s0 = {}, s1 = {};
    s0 = wmma_f16(qf0, load_B_frag(ktile[cur], HD, 0, 0, lane), s0);
    s0 = wmma_f16(qf1, load_B_frag(ktile[cur], HD, 0, 32, lane), s0);
    s1 = wmma_f16(qf0, load_B_frag(ktile[cur], HD, 16, 0, lane), s1);
    s1 = wmma_f16(qf1, load_B_frag(ktile[cur], HD, 16, 32, lane), s1);

    // Online softmax: a row of the C layout lives in one 16-lane half.
    float p0[8], p1[8], rescale[8];
#pragma unroll
    for (int j = 0; j < 8; ++j) {
      float loc = fmaxf(s0[j], s1[j]);
#pragma unroll
      for (int msk = 8; msk >= 1; msk >>= 1)
        loc = fmaxf(loc, __shfl_xor(loc, msk, 32));
      float mn = fmaxf(mrun[j], loc);
      rescale[j] = __expf(mrun[j] - mn);
      p0[j] = __expf(s0[j] - mn);
      p1[j] = __expf(s1[j] - mn);
      float rs = p0[j] + p1[j];
#pragma unroll
      for (int msk = 8; msk >= 1; msk >>= 1) rs += __shfl_xor(rs, msk, 32);
      lrun[j] = lrun[j] * rescale[j] + rs;
      mrun[j] = mn;
    }
#pragma unroll
    for (int t = 0; t < 4; ++t)
#pragma unroll
      for (int j = 0; j < 8; ++j) accO[t][j] *= rescale[j];

    // Bounce P (16x32 f16) through LDS: C layout -> A fragment layout.
#pragma unroll
    for (int j = 0; j < 8; ++j) {
      int m = j + 8 * g;
      pb[m * 32 + n] = (_Float16)p0[j];
      pb[m * 32 + 16 + n] = (_Float16)p1[j];
    }
    __syncthreads();
    v16h pf = load_A_frag(pb, 32, 0, 0, lane);

    // O += P * V : V tile in LDS, 4 d-tiles of 16
#pragma unroll
    for (int t = 0; t < 4; ++t)
      accO[t] = wmma_f16(pf, load_B_frag(vtile[cur], 32, t * 16, 0, lane),
                         accO[t]);

    async_wait_all();  // next K/V buffers staged
    __syncthreads();   // all waves done with cur buffers + pshare
  }

  // Normalize and write f16 [B,N,C] for the projection GEMM.
  const int bi = bh >> 4, h = bh & 15;
#pragma unroll
  for (int j = 0; j < 8; ++j) {
    float inv = 1.f / lrun[j];
    _Float16* dst =
        aout + ((size_t)bi * NSEQ + m0 + j + 8 * g) * CDIM + h * HD + n;
#pragma unroll
    for (int t = 0; t < 4; ++t) dst[t * 16] = (_Float16)(accO[t][j] * inv);
  }
}

// ---------------------------------------------------------------------------
// Kernel 4: output projection GEMM (same blocking as QKV) + bias, f32 out.
// ---------------------------------------------------------------------------
__global__ void __launch_bounds__(128) proj_wmma_kernel(
    const _Float16* __restrict__ ah, const _Float16* __restrict__ wproj_h,
    const float* __restrict__ b_proj, float* __restrict__ out) {
  __shared__ __align__(32) _Float16 btile[2][64 * 32];
  const int tid = threadIdx.x, lane = tid & 31, wid = tid >> 5;
  const int o0 = blockIdx.x * 64;
  const int mbase = blockIdx.y * 256 + wid * 64;
  const int g = lane >> 4, n = lane & 15;

  auto stage = [&](int k0, int buf) {
#pragma unroll
    for (int r = 0; r < 2; ++r) {
      int c = tid + 128 * r;
      int row = c >> 2, sub = c & 3;
      cp16_g2l(&btile[buf][row * 32 + sub * 8],
               wproj_h + (size_t)(o0 + row) * CDIM + k0 + sub * 8);
    }
  };

  v8f acc[4][4] = {};
  stage(0, 0);
  async_wait_all();
  __syncthreads();
  for (int k0 = 0; k0 < CDIM; k0 += 32) {
    const int cur = (k0 >> 5) & 1;
    if (k0 + 32 < CDIM) stage(k0 + 32, cur ^ 1);

    v16h af[4], bf[4];
#pragma unroll
    for (int i = 0; i < 4; ++i)
      af[i] = load_A_frag(ah, CDIM, mbase + 16 * i, k0, lane);
#pragma unroll
    for (int j = 0; j < 4; ++j)
      bf[j] = load_B_frag(btile[cur], 32, 16 * j, 0, lane);
#pragma unroll
    for (int i = 0; i < 4; ++i)
#pragma unroll
      for (int j = 0; j < 4; ++j)
        acc[i][j] = wmma_f16(af[i], bf[j], acc[i][j]);

    async_wait_all();
    __syncthreads();
  }

#pragma unroll
  for (int jt = 0; jt < 4; ++jt) {
    const int o = o0 + 16 * jt + n;
    const float bias = b_proj[o];
#pragma unroll
    for (int it = 0; it < 4; ++it) {
      float* dst = out + (size_t)(mbase + 16 * it + 8 * g) * CDIM + o;
#pragma unroll
      for (int jj = 0; jj < 8; ++jj)
        dst[(size_t)jj * CDIM] = acc[it][jt][jj] + bias;
    }
  }
}

// ---------------------------------------------------------------------------
extern "C" void kernel_launch(void* const* d_in, const int* in_sizes, int n_in,
                              void* d_out, int out_size, void* d_ws,
                              size_t ws_size, hipStream_t stream) {
  (void)in_sizes; (void)n_in; (void)out_size; (void)ws_size;
  const float* x      = (const float*)d_in[0];
  const float* W_qkv  = (const float*)d_in[1];
  const float* b_qkv  = (const float*)d_in[2];
  const float* lora_A = (const float*)d_in[3];
  const float* lora_B = (const float*)d_in[4];
  const float* W_proj = (const float*)d_in[5];
  const float* b_proj = (const float*)d_in[6];
  // d_in[7] = idx : single adapter modeled, unused.

  char* ws = (char*)d_ws;
  size_t off = 0;
  auto alloc = [&](size_t bytes) -> void* {
    void* p = ws + off;
    off = (off + bytes + 255) & ~(size_t)255;
    return p;
  };
  _Float16* xh      = (_Float16*)alloc((size_t)MROWS * CDIM * 2);
  _Float16* wqkv_h  = (_Float16*)alloc((size_t)OUTQKV * CDIM * 2);
  _Float16* wproj_h = (_Float16*)alloc((size_t)CDIM * CDIM * 2);
  float*    abuf    = (float*)   alloc((size_t)MROWS * 8 * 4);
  _Float16* qbuf    = (_Float16*)alloc((size_t)BB * NHEADS * NSEQ * HD * 2);
  _Float16* kbuf    = (_Float16*)alloc((size_t)BB * NHEADS * NSEQ * HD * 2);
  _Float16* vtbuf   = (_Float16*)alloc((size_t)BB * NHEADS * HD * NSEQ * 2);
  _Float16* aout    = (_Float16*)alloc((size_t)MROWS * CDIM * 2);

  // 0) f32 -> f16 conversions
  cvt_f32_to_f16<<<4096, 256, 0, stream>>>(x, xh, MROWS * CDIM);
  cvt_f32_to_f16<<<4096, 256, 0, stream>>>(W_qkv, wqkv_h, OUTQKV * CDIM);
  cvt_f32_to_f16<<<2048, 256, 0, stream>>>(W_proj, wproj_h, CDIM * CDIM);

  // 1) LoRA a = x . lora_A^T
  lora_a_kernel<<<(MROWS * 2 * RLORA + 255) / 256, 256, 0, stream>>>(x, lora_A,
                                                                     abuf);

  // 2) fused QKV GEMM + bias + LoRA, pack q/k/vT f16
  qkv_wmma_kernel<<<dim3(OUTQKV / 64, MROWS / 256), 128, 0, stream>>>(
      xh, wqkv_h, b_qkv, lora_B, abuf, qbuf, kbuf, vtbuf);

  // 3) flash attention: 32 (b,h) x 32 query groups, 4 waves/block share K/V
  flash_attn_kernel<<<32 * 32, 128, 0, stream>>>(qbuf, kbuf, vtbuf, aout);

  // 4) projection GEMM + bias -> f32 output
  proj_wmma_kernel<<<dim3(CDIM / 64, MROWS / 256), 128, 0, stream>>>(
      aout, wproj_h, b_proj, (float*)d_out);
}
